// DyGFormer_45260365365904
// MI455X (gfx1250) — matrix operand order repaired and checked by reading, not compiled, not run on hardware
//
#include <hip/hip_runtime.h>

#define F 128
#define L 512
#define BATCH 256
#define NBINS 1024
#define TROWS 528   // 513 possible count values (0..512) padded to multiple of 16

typedef __attribute__((ext_vector_type(2)))  float    v2f;
typedef __attribute__((ext_vector_type(8)))  float    v8f;
typedef __attribute__((ext_vector_type(16))) _Float16 v16h;

// ---------------------------------------------------------------------------
// Kernel 1: H[v,f] = relu(v * W1[f] + b1[f]) for v in [0, TROWS), f in [0, F)
// ---------------------------------------------------------------------------
__global__ __launch_bounds__(256)
void k_build_h(const float* __restrict__ W1, const float* __restrict__ b1,
               float* __restrict__ H) {
    int idx = blockIdx.x * 256 + threadIdx.x;
    if (idx >= TROWS * F) return;
    int v = idx >> 7;         // / F
    int f = idx & (F - 1);
    float h = fmaf((float)v, W1[f], b1[f]);
    H[idx] = h > 0.0f ? h : 0.0f;
}

// ---------------------------------------------------------------------------
// Kernel 2: T[v,g] = sum_f H[v,f] * W2[g,f]   (528x128x128 GEMM via WMMA f32)
// One wave (32 threads) per 16x16 output tile; 33 x 8 = 264 tiles.
// ---------------------------------------------------------------------------
__global__ __launch_bounds__(32)
void k_table_gemm(const float* __restrict__ H, const float* __restrict__ W2,
                  float* __restrict__ T) {
    const int tile = blockIdx.x;
    const int m0   = (tile >> 3) << 4;   // row-tile base (count value v)
    const int n0   = (tile & 7)  << 4;   // col-tile base (output feature g)
    const int lane = threadIdx.x;
    const int r    = lane & 15;
    const int half = lane >> 4;          // 0: lanes 0-15, 1: lanes 16-31

    v8f acc = {};

#if __has_builtin(__builtin_amdgcn_wmma_f32_16x16x4_f32)
    // Native fp32 WMMA: D(16x16) += A(16x4) x B(4x16), K stepped by 4.
    // A layout: lane r holds A[m=r][k0+2*half], A[m=r][k0+2*half+1]
    // B layout (mirror): lane r holds B[k0+2*half][n=r], B[k0+2*half+1][n=r]
    //                    with B[k][n] = W2[n0+n][k]  (B = W2^T tile)
    for (int k0 = 0; k0 < F; k0 += 4) {
        const int ka = k0 + 2 * half;
        v2f a, b;
        a.x = H[(m0 + r) * F + ka];
        a.y = H[(m0 + r) * F + ka + 1];
        b.x = W2[(n0 + r) * F + ka];
        b.y = W2[(n0 + r) * F + ka + 1];
        acc = __builtin_amdgcn_wmma_f32_16x16x4_f32(
            /*neg_a=*/false, a, /*neg_b=*/false, b,
            /*c_mod=*/(short)0, acc, /*reuse_a=*/false, /*reuse_b=*/false);
    }
#else
    // Fallback: codegen-confirmed fp16 WMMA 16x16x32 (convert on the fly).
    for (int k0 = 0; k0 < F; k0 += 32) {
        v16h a, b;
        #pragma unroll
        for (int j = 0; j < 16; ++j) {
            int ka = k0 + (j & 7) + ((j >> 3) << 4) + (half << 3);
            a[j] = (_Float16)H[(m0 + r) * F + ka];
            int kb = k0 + (j & 7) + ((j >> 3) << 4) + (half << 3);
            b[j] = (_Float16)W2[(n0 + r) * F + kb];
        }
        acc = __builtin_amdgcn_wmma_f32_16x16x32_f16(
            false, a, false, b, (short)0, acc, false, false);
    }
#endif

    // C/D layout: VGPR i -> row m0+i (lanes 0-15) / m0+8+i (lanes 16-31)
    #pragma unroll
    for (int i = 0; i < 8; ++i) {
        int row = m0 + i + (half << 3);
        T[row * F + n0 + r] = acc[i];
    }
}

// ---------------------------------------------------------------------------
// Kernel 3: per-row LDS histograms -> count lookup -> table gather -> stream out
// 4 blocks per batch row; each 32-lane group emits full 128-float rows (float4).
// out[b,l,:] = T[c_own] + T[c_other] + 2*b2
// ---------------------------------------------------------------------------
__global__ __launch_bounds__(256)
void k_count_encode(const int* __restrict__ src_ids, const int* __restrict__ dst_ids,
                    const float* __restrict__ T, const float* __restrict__ b2,
                    float* __restrict__ out_src, float* __restrict__ out_dst) {
    __shared__ int hist_s[NBINS];
    __shared__ int hist_d[NBINS];
    __shared__ __align__(16) float b2x2[F];

    const int b     = blockIdx.x >> 2;   // batch row
    const int chunk = blockIdx.x & 3;    // quarter of the L dimension
    const int tid   = threadIdx.x;

    for (int i = tid; i < NBINS; i += 256) { hist_s[i] = 0; hist_d[i] = 0; }
    if (tid < F) b2x2[tid] = 2.0f * b2[tid];
    __syncthreads();

    const int* __restrict__ srow = src_ids + b * L;
    const int* __restrict__ drow = dst_ids + b * L;
    for (int i = tid; i < L; i += 256) {
        atomicAdd(&hist_s[srow[i] & (NBINS - 1)], 1);
        atomicAdd(&hist_d[drow[i] & (NBINS - 1)], 1);
    }
    __syncthreads();

    const int lane = tid & 31;           // 32 lanes x float4 = 128 floats/row
    const int grp  = tid >> 5;           // 8 groups, each owns an l at a time
    const float4 bb = reinterpret_cast<const float4*>(b2x2)[lane];

    const int l_lo = chunk * (L / 4);
    const int l_hi = l_lo + (L / 4);
    for (int l = l_lo + grp; l < l_hi; l += 8) {
        const int sid = srow[l] & (NBINS - 1);
        const int did = drow[l] & (NBINS - 1);
        int css = 0, csd = 0, cds = 0, cdd = 0;
        if (sid != 0) { css = hist_s[sid]; csd = hist_d[sid]; }   // masked -> 0
        if (did != 0) { cds = hist_s[did]; cdd = hist_d[did]; }

        const float4 t_ss = reinterpret_cast<const float4*>(T + css * F)[lane];
        const float4 t_sd = reinterpret_cast<const float4*>(T + csd * F)[lane];
        const float4 t_ds = reinterpret_cast<const float4*>(T + cds * F)[lane];
        const float4 t_dd = reinterpret_cast<const float4*>(T + cdd * F)[lane];

        float4 os, od;
        os.x = t_ss.x + t_sd.x + bb.x;  os.y = t_ss.y + t_sd.y + bb.y;
        os.z = t_ss.z + t_sd.z + bb.z;  os.w = t_ss.w + t_sd.w + bb.w;
        od.x = t_ds.x + t_dd.x + bb.x;  od.y = t_ds.y + t_dd.y + bb.y;
        od.z = t_ds.z + t_dd.z + bb.z;  od.w = t_ds.w + t_dd.w + bb.w;

        const size_t row_off = (size_t)(b * L + l) * F;
        reinterpret_cast<float4*>(out_src + row_off)[lane] = os;
        reinterpret_cast<float4*>(out_dst + row_off)[lane] = od;
    }
}

// ---------------------------------------------------------------------------
extern "C" void kernel_launch(void* const* d_in, const int* in_sizes, int n_in,
                              void* d_out, int out_size, void* d_ws, size_t ws_size,
                              hipStream_t stream) {
    (void)in_sizes; (void)n_in; (void)out_size; (void)ws_size;

    const int*   src_ids = (const int*)  d_in[0];
    const int*   dst_ids = (const int*)  d_in[1];
    const float* W1      = (const float*)d_in[2];   // (F,1)
    const float* b1      = (const float*)d_in[3];   // (F,)
    const float* W2      = (const float*)d_in[4];   // (F,F)
    const float* b2      = (const float*)d_in[5];   // (F,)

    float* out_src = (float*)d_out;
    float* out_dst = out_src + (size_t)BATCH * L * F;

    float* H = (float*)d_ws;           // TROWS x F  (~270 KB)
    float* T = H + TROWS * F;          // TROWS x F  (~270 KB)

    k_build_h<<<(TROWS * F + 255) / 256, 256, 0, stream>>>(W1, b1, H);
    k_table_gemm<<<(TROWS / 16) * (F / 16), 32, 0, stream>>>(H, W2, T);
    k_count_encode<<<BATCH * 4, 256, 0, stream>>>(src_ids, dst_ids, T, b2,
                                                  out_src, out_dst);
}
